// NNCRF_79989470920843
// MI455X (gfx1250) — matrix operand order repaired and checked by reading, not compiled, not run on hardware
//
#include <hip/hip_runtime.h>
#include <hip/hip_bf16.h>
#include <math.h>

// ---------------- problem constants ----------------
#define T_LEN 512
#define BATCH 64
#define D_IN  768
#define HID   384
#define GATE  1536              // 4*HID
#define KTAGS 25
#define M_ROWS (T_LEN*BATCH)    // 32768 rows (t*B+b)
#define KDIM  768               // contraction dim for both layers (D and 2H)
#define KT_CH (KDIM/32)         // 24 k-chunks for input GEMM
#define NT_TOT (GATE/16)        // 96 n-tiles
#define HKT   (HID/32)          // 12 k-chunks for recurrent GEMM

typedef __bf16 bf16_t;
typedef __attribute__((ext_vector_type(16))) __bf16 v16bf;
typedef __attribute__((ext_vector_type(8)))  float  f32x8;

union FragU { uint4 q[2]; unsigned int u[8]; v16bf v; };
union PackU { bf16_t h[2]; unsigned int u; };

__device__ __forceinline__ float sigf(float x) { return 1.f / (1.f + __expf(-x)); }

// ================= fragment packing =================
// A-fragment (16x32 bf16, ISA layout): lane L: m=L&15, half=L>>4;
// dword i: K = (i<4 ? half*8+2i : 16+half*8+2(i-4)) (+0,+1 packed)
__device__ __forceinline__ void a_decode(long id, long& row, int& k) {
  int i = id & 7; int lane = (id >> 3) & 31; long tile = id >> 8;
  int kt = (int)(tile % KT_CH); long mt = tile / KT_CH;
  int half = lane >> 4, mr = lane & 15;
  int kb = (i < 4) ? (half * 8 + 2 * i) : (16 + half * 8 + 2 * (i - 4));
  row = mt * 16 + mr;
  k = kt * 32 + kb;
}

// Layer-0 A: word_embeds (B,T,D) f32  ->  packed rows (t*B+b)
__global__ void pack_embeds(const float* __restrict__ we, unsigned int* __restrict__ ap) {
  long id = (long)blockIdx.x * blockDim.x + threadIdx.x;
  if (id >= (long)M_ROWS * KDIM / 2) return;
  long row; int k; a_decode(id, row, k);
  int t = (int)(row / BATCH), b = (int)(row % BATCH);
  const float* s = we + ((long)b * T_LEN + t) * D_IN + k;
  PackU pk; pk.h[0] = (bf16_t)s[0]; pk.h[1] = (bf16_t)s[1];
  ap[id] = pk.u;
}

// Layer-1 A: Hout (rows t*B+b, 768 cols) bf16 -> packed
__global__ void pack_h(const bf16_t* __restrict__ hsrc, unsigned int* __restrict__ ap) {
  long id = (long)blockIdx.x * blockDim.x + threadIdx.x;
  if (id >= (long)M_ROWS * KDIM / 2) return;
  long row; int k; a_decode(id, row, k);
  const bf16_t* s = hsrc + row * (2 * HID) + k;
  PackU pk; pk.h[0] = s[0]; pk.h[1] = s[1];
  ap[id] = pk.u;
}

// B-fragment (32x16 bf16): lane L: n=L&15, halfK=L>>4; dword j: K=(j<4 ? halfK*8+2j : 16+halfK*8+2(j-4)) (+0,+1)
// Source weight W is (N rows, K cols) row-major f32, B[k][n] = W[n][k]  (i.e. W^T).
__global__ void pack_w(const float* __restrict__ w, unsigned int* __restrict__ bp,
                       int N, int K) {
  long id = (long)blockIdx.x * blockDim.x + threadIdx.x;
  if (id >= (long)N * K / 2) return;
  int j = id & 7; int lane = (id >> 3) & 31; long tile = id >> 8;
  int NTT = N / 16;
  int nt = (int)(tile % NTT); int kt = (int)(tile / NTT);
  int n = nt * 16 + (lane & 15);
  int k = kt * 32 + ((j < 4) ? (((lane >> 4) << 3) + 2 * j) : (16 + ((lane >> 4) << 3) + 2 * (j - 4)));
  PackU pk;
  pk.h[0] = (bf16_t)w[(long)n * K + k];
  pk.h[1] = (bf16_t)w[(long)n * K + k + 1];
  bp[id] = pk.u;
}

// ================= big input GEMM: XG = X @ Wih^T + (bih+bhh) =================
// M=32768, N=1536, K=768.  8 waves/block; each wave: one m-tile x 4 n-tiles.
// Software-pipelined: A and the 4 B fragments are double-buffered across kt so
// next chunk's global_load_b128s overlap the current chunk's 4 WMMAs.
__global__ void __launch_bounds__(256)
gemm_gates(const unsigned int* __restrict__ Ap, const unsigned int* __restrict__ Bp,
           const float* __restrict__ bih, const float* __restrict__ bhh,
           bf16_t* __restrict__ XG) {
  const int NG = NT_TOT / 4;                       // 24 n-groups
  int lane = threadIdx.x & 31, wave = threadIdx.x >> 5;
  long job = (long)blockIdx.x * 8 + wave;          // jobs = 2048*24 = 49152
  int ng = (int)(job % NG);
  long mt = job / NG;
  if (mt >= M_ROWS / 16) return;

  const unsigned int* aBase = Ap + ((mt * KT_CH) * 32 + lane) * 8;
  const unsigned int* bBase = Bp + ((long)(ng * 4) * 32 + lane) * 8;
  // B fragments for chunk kt, gate g live at bBase + kt*96*256 + g*256 dwords.

  f32x8 zero = {0.f, 0.f, 0.f, 0.f, 0.f, 0.f, 0.f, 0.f};
  f32x8 acc[4] = {zero, zero, zero, zero};

  FragU fa[2];
  FragU fb[2][4];

  // prologue: fill buffer 0
  {
    const uint4* pa = (const uint4*)aBase;
    fa[0].q[0] = pa[0]; fa[0].q[1] = pa[1];
#pragma unroll
    for (int g = 0; g < 4; ++g) {
      const uint4* pb = (const uint4*)(bBase + g * 256);
      fb[0][g].q[0] = pb[0]; fb[0][g].q[1] = pb[1];
    }
  }

  for (int kt = 0; kt < KT_CH; ++kt) {
    int cur = kt & 1, nxt = cur ^ 1;
    if (kt + 1 < KT_CH) {                          // issue next chunk's loads first
      const uint4* pa = (const uint4*)(aBase + (long)(kt + 1) * 256);
      fa[nxt].q[0] = pa[0]; fa[nxt].q[1] = pa[1];
      const unsigned int* bn = bBase + (long)(kt + 1) * (NT_TOT * 256);
#pragma unroll
      for (int g = 0; g < 4; ++g) {
        const uint4* pb = (const uint4*)(bn + g * 256);
        fb[nxt][g].q[0] = pb[0]; fb[nxt][g].q[1] = pb[1];
      }
    }
    if (kt + 2 < KT_CH) {                          // gfx1250 global_prefetch_b8 path
      __builtin_prefetch(aBase + (long)(kt + 2) * 256, 0, 3);
      __builtin_prefetch(bBase + (long)(kt + 2) * (NT_TOT * 256), 0, 3);
    }
#pragma unroll
    for (int g = 0; g < 4; ++g)
      acc[g] = __builtin_amdgcn_wmma_f32_16x16x32_bf16(
          false, fa[cur].v, false, fb[cur][g].v, (short)0, acc[g], false, false);
  }

  int hi8 = (lane >> 4) << 3, nc = lane & 15;
#pragma unroll
  for (int g = 0; g < 4; ++g) {
    int col = (ng * 4 + g) * 16 + nc;
    float bs = bih[col] + bhh[col];
#pragma unroll
    for (int r = 0; r < 8; ++r) {
      long row = mt * 16 + r + hi8;                // row = t*B + b
      XG[row * GATE + col] = (bf16_t)(acc[g][r] + bs);
    }
  }
}

// ================= persistent LSTM recurrence =================
// grid.x = 2 (dir: 0=fwd, 1=bwd), 512 threads = 16 waves.
// h (bf16, padded) lives in LDS as the WMMA A operand; c lives in registers
// (each gate element has a unique owner thread across all 512 steps).
__global__ void __launch_bounds__(512)
lstm_seq(const bf16_t* __restrict__ XGf, const bf16_t* __restrict__ XGr,
         const unsigned int* __restrict__ WPf, const unsigned int* __restrict__ WPr,
         const int* __restrict__ lens, bf16_t* __restrict__ Hout) {
  const int HSTR = HID + 2;                        // pad: kill LDS bank conflicts
  __shared__ bf16_t h_s[BATCH * (HID + 2)];
  __shared__ int lens_s[BATCH];

  int dir = blockIdx.x;
  const bf16_t* XG = dir ? XGr : XGf;
  const unsigned int* WP = dir ? WPr : WPf;
  int lane = threadIdx.x & 31, wave = threadIdx.x >> 5;
  int half = lane >> 4, mr = lane & 15, hi8 = half << 3, nc = lane & 15;

  for (int i = threadIdx.x; i < BATCH * HSTR; i += blockDim.x) h_s[i] = (bf16_t)0.f;
  if (threadIdx.x < BATCH) lens_s[threadIdx.x] = lens[threadIdx.x];
  __syncthreads();

  float cst[6][8];   // cell state (owned exclusively by this thread)
  float hst[6][8];   // staged h_new (written to LDS after barrier)
#pragma unroll
  for (int q = 0; q < 6; ++q)
#pragma unroll
    for (int r = 0; r < 8; ++r) cst[q][r] = 0.f;

  for (int t = 0; t < T_LEN; ++t) {
#pragma unroll
    for (int q = 0; q < 6; ++q) {
      int job = wave * 6 + q;                      // 96 jobs = 4 m-tiles x 24 j-tiles
      int mt = job & 3, jt = job >> 2;
      f32x8 acc[4];
      // init accumulators with precomputed input gates (bwd reads reversed rows)
#pragma unroll
      for (int g = 0; g < 4; ++g) {
        int col = g * HID + jt * 16 + nc;
#pragma unroll
        for (int r = 0; r < 8; ++r) {
          int b = mt * 16 + r + hi8;
          int len = lens_s[b];
          int rt = dir ? ((t < len) ? (len - 1 - t) : t) : t;
          acc[g][r] = (float)XG[((long)rt * BATCH + b) * GATE + col];
        }
      }
      // accumulate h @ Whh^T : A from LDS, B fragments from L2-resident pack.
      // All 4 gate B-fragments are loaded before the 4 WMMAs so the 8
      // global_load_b128s overlap the XDL work.
      const unsigned int* wBase = WP + ((long)jt * 32 + lane) * 8;
      for (int kt = 0; kt < HKT; ++kt) {
        FragU fb[4];
        const unsigned int* wk = wBase + (long)kt * (NT_TOT * 256);
#pragma unroll
        for (int g = 0; g < 4; ++g) {
          const uint4* pb = (const uint4*)(wk + (long)g * (HID / 16) * 256);
          fb[g].q[0] = pb[0]; fb[g].q[1] = pb[1];
        }
        FragU fa;
        int ktb = kt * 32, m = mt * 16 + mr;
#pragma unroll
        for (int i = 0; i < 8; ++i) {
          int kb = (i < 4) ? (hi8 + (i << 1)) : (16 + hi8 + ((i - 4) << 1));
          fa.u[i] = *(const unsigned int*)&h_s[m * HSTR + ktb + kb];
        }
#pragma unroll
        for (int g = 0; g < 4; ++g)
          acc[g] = __builtin_amdgcn_wmma_f32_16x16x32_bf16(
              false, fa.v, false, fb[g].v, (short)0, acc[g], false, false);
      }
      // pointwise LSTM + masking (C/D layout: n = lane&15, m = r + 8*(lane>=16))
#pragma unroll
      for (int r = 0; r < 8; ++r) {
        int b = mt * 16 + r + hi8;
        int len = lens_s[b];
        bool mv = (t < len);
        float iv = acc[0][r], fv = acc[1][r], gv = acc[2][r], ov = acc[3][r];
        float c_old = cst[q][r];
        float cn = sigf(fv) * c_old + sigf(iv) * tanhf(gv);
        float hn = sigf(ov) * tanhf(cn);
        cst[q][r] = mv ? cn : c_old;
        int col = jt * 16 + nc;
        float h_old = (float)h_s[b * HSTR + col];
        hst[q][r] = mv ? hn : h_old;
        int rt = dir ? ((t < len) ? (len - 1 - t) : t) : t;   // out un-reversal
        Hout[((long)rt * BATCH + b) * (2 * HID) + dir * HID + col] =
            (bf16_t)(mv ? hn : 0.f);
      }
    }
    __syncthreads();                               // all h reads done
#pragma unroll
    for (int q = 0; q < 6; ++q) {
      int job = wave * 6 + q;
      int mt = job & 3, jt = job >> 2;
#pragma unroll
      for (int r = 0; r < 8; ++r) {
        int b = mt * 16 + r + hi8;
        h_s[b * HSTR + jt * 16 + nc] = (bf16_t)hst[q][r];
      }
    }
    __syncthreads();                               // new h visible
  }
}

// ================= emissions: em = 10*(h @ W_tag^T + b_tag) =================
__global__ void tag_proj(const bf16_t* __restrict__ H2, const float* __restrict__ Wt,
                         const float* __restrict__ bt, float* __restrict__ em) {
  __shared__ float hrow[2 * HID];
  long row = blockIdx.x;
  int lane = threadIdx.x;
  for (int d = lane; d < 2 * HID; d += 32) hrow[d] = (float)H2[row * (2 * HID) + d];
  __syncthreads();
  if (lane < KTAGS) {
    float s = 0.f;
    for (int d = 0; d < 2 * HID; ++d) s += hrow[d] * Wt[lane * (2 * HID) + d];
    em[row * KTAGS + lane] = 10.f * (s + bt[lane]);
  }
}

// ================= CRF log-likelihood, one block per batch element =================
__global__ void crf_llh(const float* __restrict__ em, const int* __restrict__ tags,
                        const int* __restrict__ lens, const float* __restrict__ startT,
                        const float* __restrict__ endT, const float* __restrict__ trans,
                        float* __restrict__ llh) {
  __shared__ float tr[KTAGS * KTAGS];
  __shared__ float al[KTAGS];
  int b = blockIdx.x, lane = threadIdx.x;
  for (int i = lane; i < KTAGS * KTAGS; i += 32) tr[i] = trans[i];
  int len = lens[b];
  __syncthreads();

  float alpha = 0.f;
  if (lane < KTAGS) { alpha = startT[lane] + em[(long)b * KTAGS + lane]; al[lane] = alpha; }
  __syncthreads();

  float score = 0.f;
  if (lane == 0) {                                 // numerator (prefix mask: t < len)
    int tg = tags[b * T_LEN];
    score = startT[tg] + em[(long)b * KTAGS + tg];
    int prev = tg;
    for (int t = 1; t < len; ++t) {
      int tgt = tags[b * T_LEN + t];
      score += tr[prev * KTAGS + tgt] + em[((long)t * BATCH + b) * KTAGS + tgt];
      prev = tgt;
    }
    score += endT[tags[b * T_LEN + (len - 1)]];
  }

  for (int t = 1; t < len; ++t) {                  // forward algorithm (streaming LSE)
    float na = alpha;
    if (lane < KTAGS) {
      float mx = -1e30f, sm = 0.f;
#pragma unroll
      for (int i = 0; i < KTAGS; ++i) {
        float x = al[i] + tr[i * KTAGS + lane];
        if (x > mx) { sm = sm * __expf(mx - x) + 1.f; mx = x; }
        else        { sm += __expf(x - mx); }
      }
      na = em[((long)t * BATCH + b) * KTAGS + lane] + mx + __logf(sm);
    }
    __syncthreads();
    alpha = na;
    if (lane < KTAGS) al[lane] = alpha;
    __syncthreads();
  }

  if (lane < KTAGS) al[lane] = alpha + endT[lane];
  __syncthreads();
  if (lane == 0) {
    float mx = -1e30f, sm = 0.f;
    for (int i = 0; i < KTAGS; ++i) {
      float x = al[i];
      if (x > mx) { sm = sm * __expf(mx - x) + 1.f; mx = x; }
      else        { sm += __expf(x - mx); }
    }
    llh[b] = score - (mx + __logf(sm));
  }
}

__global__ void mean64(const float* __restrict__ llh, float* __restrict__ out) {
  __shared__ float s[64];
  s[threadIdx.x] = llh[threadIdx.x];
  __syncthreads();
  for (int st = 32; st > 0; st >>= 1) {
    if (threadIdx.x < st) s[threadIdx.x] += s[threadIdx.x + st];
    __syncthreads();
  }
  if (threadIdx.x == 0) out[0] = s[0] / 64.f;
}

// ================= host orchestration =================
extern "C" void kernel_launch(void* const* d_in, const int* in_sizes, int n_in,
                              void* d_out, int out_size, void* d_ws, size_t ws_size,
                              hipStream_t stream) {
  (void)in_sizes; (void)n_in; (void)out_size; (void)ws_size;
  const float* we    = (const float*)d_in[0];
  const int*   lens  = (const int*)d_in[1];
  const int*   tags  = (const int*)d_in[2];
  // d_in[3] = mask: recomputed from lens instead.
  const float* Wih[4]  = {(const float*)d_in[4],  (const float*)d_in[8],
                          (const float*)d_in[12], (const float*)d_in[16]};
  const float* Whh[4]  = {(const float*)d_in[5],  (const float*)d_in[9],
                          (const float*)d_in[13], (const float*)d_in[17]};
  const float* bihp[4] = {(const float*)d_in[6],  (const float*)d_in[10],
                          (const float*)d_in[14], (const float*)d_in[18]};
  const float* bhhp[4] = {(const float*)d_in[7],  (const float*)d_in[11],
                          (const float*)d_in[15], (const float*)d_in[19]};
  const float* Wt = (const float*)d_in[20];
  const float* bt = (const float*)d_in[21];
  const float* startT = (const float*)d_in[22];
  const float* endT   = (const float*)d_in[23];
  const float* trans  = (const float*)d_in[24];

  size_t off = 0;
  auto take = [&](size_t bytes) -> void* {
    off = (off + 255) & ~(size_t)255;
    void* p = (char*)d_ws + off;
    off += bytes;
    return p;
  };
  unsigned int* Apack = (unsigned int*)take((size_t)M_ROWS * KDIM * 2);   // 50 MB
  bf16_t* XGf = (bf16_t*)take((size_t)M_ROWS * GATE * 2);                 // 100 MB
  bf16_t* XGr = (bf16_t*)take((size_t)M_ROWS * GATE * 2);                 // 100 MB
  bf16_t* Hout = (bf16_t*)take((size_t)M_ROWS * 2 * HID * 2);             // 50 MB
  unsigned int* WihP[4]; unsigned int* WhhP[4];
  for (int i = 0; i < 4; ++i) {
    WihP[i] = (unsigned int*)take((size_t)KDIM * GATE * 2);
    WhhP[i] = (unsigned int*)take((size_t)HID * GATE * 2);
  }
  float* em  = (float*)take((size_t)M_ROWS * KTAGS * 4);
  float* llh = (float*)take(64 * 4);

  for (int i = 0; i < 4; ++i) {
    pack_w<<<(GATE * KDIM / 2) / 256, 256, 0, stream>>>(Wih[i], WihP[i], GATE, KDIM);
    pack_w<<<(GATE * HID  / 2) / 256, 256, 0, stream>>>(Whh[i], WhhP[i], GATE, HID);
  }
  pack_embeds<<<(M_ROWS * KDIM / 2) / 256, 256, 0, stream>>>(we, Apack);

  for (int l = 0; l < 2; ++l) {
    if (l == 1)
      pack_h<<<(M_ROWS * KDIM / 2) / 256, 256, 0, stream>>>(Hout, Apack);
    int iF = 2 * l, iR = 2 * l + 1;
    const int gblocks = (M_ROWS / 16) * (NT_TOT / 4) / 8;   // 6144
    gemm_gates<<<gblocks, 256, 0, stream>>>(Apack, WihP[iF], bihp[iF], bhhp[iF], XGf);
    gemm_gates<<<gblocks, 256, 0, stream>>>(Apack, WihP[iR], bihp[iR], bhhp[iR], XGr);
    lstm_seq<<<2, 512, 0, stream>>>(XGf, XGr, WhhP[iF], WhhP[iR], lens, Hout);
  }

  tag_proj<<<M_ROWS, 32, 0, stream>>>(Hout, Wt, bt, em);
  crf_llh<<<BATCH, 32, 0, stream>>>(em, tags, lens, startT, endT, trans, llh);
  mean64<<<1, 64, 0, stream>>>(llh, (float*)d_out);
}
